// Attention_26517128085974
// MI455X (gfx1250) — matrix operand order repaired and checked by reading, not compiled, not run on hardware
//
#include <hip/hip_runtime.h>

typedef __attribute__((ext_vector_type(16))) __bf16 v16bf;
typedef __attribute__((ext_vector_type(16))) unsigned short v16us;
typedef __attribute__((ext_vector_type(8))) float v8f;
typedef __attribute__((ext_vector_type(4))) unsigned int v4u;
typedef __attribute__((ext_vector_type(8))) int v8i;
typedef __attribute__((ext_vector_type(4))) int v4i;

__device__ __forceinline__ unsigned short f2bf(float f) {
  unsigned int u = __builtin_bit_cast(unsigned int, f);
  unsigned int r = u + 0x7FFFu + ((u >> 16) & 1u);   // round-to-nearest-even
  return (unsigned short)(r >> 16);
}
__device__ __forceinline__ float bf2f(unsigned short s) {
  unsigned int u = ((unsigned int)s) << 16;
  return __builtin_bit_cast(float, u);
}
__device__ __forceinline__ unsigned int pk2bf(float lo, float hi) {
  return (unsigned int)f2bf(lo) | ((unsigned int)f2bf(hi) << 16);
}
__device__ __forceinline__ void store_out(float* p, float v) { *p = v; }
__device__ __forceinline__ void store_out(unsigned short* p, float v) { *p = f2bf(v); }

// helpers so the A-stager works for both fp32 and bf16 sources
// (must be declared BEFORE the template that calls them: builtin arg types
//  have no ADL, so only definition-time lookup applies)
__device__ __forceinline__ float bf2f_or_id(float f) { return f; }
__device__ __forceinline__ float bf2f_or_id(unsigned short s) { return bf2f(s); }

__device__ __forceinline__ v8f wmma_bf16(const v16us& a, const v16us& b, v8f c) {
  return __builtin_amdgcn_wmma_f32_16x16x32_bf16(
      false, __builtin_bit_cast(v16bf, a),
      false, __builtin_bit_cast(v16bf, b),
      (short)0, c, false, false);
}

// LDS byte offset of a __shared__ object (ptrtoint of an addrspace(3) pointer)
typedef __attribute__((address_space(3))) unsigned short lds_us;
__device__ __forceinline__ unsigned lds_off(unsigned short* p) {
  return (unsigned)(unsigned long long)(lds_us*)p;
}

// Tensor Data Mover: 2D bf16 tile load, toolchain-arity hedge.
__device__ __forceinline__ void tdm_load_2d(v4u g0, v8i g1) {
  v4i z4 = {0, 0, 0, 0};
#if __clang_major__ >= 23
  v8i z8 = {0, 0, 0, 0, 0, 0, 0, 0};
  __builtin_amdgcn_tensor_load_to_lds(g0, g1, z4, z4, z8, 0);
#else
  __builtin_amdgcn_tensor_load_to_lds(g0, g1, z4, z4, 0);
#endif
}

// ---------------------------------------------------------------------------
// GEMM: C[M,N] = A[M,K] * W[K,N] + bias, bf16 WMMA, fp32 accumulate.
// Block tile 128x64, BK=32. 8 waves (4x2), each wave = 32x32 (2x2 WMMA tiles).
// ---------------------------------------------------------------------------
template <typename AT, typename OT>
__global__ __launch_bounds__(256) void gemm_bias_wmma(
    const AT* __restrict__ A, const float* __restrict__ W,
    const float* __restrict__ bias, OT* __restrict__ Cout,
    int M, int N, int K) {
  constexpr int BM = 128, BN = 64, BK = 32;
  constexpr int AS = 40;  // halfword stride (80B, 16B aligned)
  constexpr int BS = 40;
  __shared__ unsigned short Asm[BM * AS];  // [m][k] bf16
  __shared__ unsigned short Bsm[BN * BS];  // [n][k] bf16 (W transposed)
  unsigned int* As32 = (unsigned int*)Asm;
  unsigned int* Bs32 = (unsigned int*)Bsm;

  const int tid = threadIdx.x;
  const int lane = tid & 31;
  const int w = tid >> 5;
  const int wm = w >> 1, wn = w & 1;
  const int l15 = lane & 15;
  const int halfsel = (lane < 16) ? 0 : 1;
  const int m0 = blockIdx.y * BM, n0 = blockIdx.x * BN;

  v8f acc[2][2] = {};

  const int arow = tid >> 1;            // 0..127
  const int acol = (tid & 1) * 16;      // 0 / 16
  const int kpair = (tid & 15) * 2;     // k: 0,2,..,30
  const int ncol = (tid >> 4) * 4;      // n: 0..60

  for (int k0 = 0; k0 < K; k0 += BK) {
    __syncthreads();
    {  // stage A tile -> bf16, packed 32-bit LDS stores
      const AT* ap = A + (size_t)(m0 + arow) * K + k0 + acol;
#pragma unroll
      for (int j = 0; j < 8; ++j)
        As32[arow * (AS / 2) + (acol >> 1) + j] =
            pk2bf(bf2f_or_id(ap[2 * j]), bf2f_or_id(ap[2 * j + 1]));
      if (k0 + BK < K) __builtin_prefetch(ap + BK, 0, 1);
    }
    {  // stage W tile transposed [n][k]: 2 k-rows x 4 n per thread, pack along k
      const float* wp0 = W + (size_t)(k0 + kpair) * N + n0 + ncol;
      const float* wp1 = wp0 + N;
#pragma unroll
      for (int j = 0; j < 4; ++j)
        Bs32[(ncol + j) * (BS / 2) + (kpair >> 1)] = pk2bf(wp0[j], wp1[j]);
      if (k0 + BK < K) __builtin_prefetch(wp0 + (size_t)BK * N, 0, 1);
    }
    __syncthreads();

    v16us af[2], bfr[2];
#pragma unroll
    for (int mt = 0; mt < 2; ++mt) {
      const int row = wm * 32 + mt * 16 + l15;
      const int kb = halfsel * 16;
#pragma unroll
      for (int e = 0; e < 16; ++e)
        af[mt][e] = Asm[row * AS + e + kb];
    }
#pragma unroll
    for (int nt = 0; nt < 2; ++nt) {
      const int col = wn * 32 + nt * 16 + l15;
      const int kb = halfsel * 16;
#pragma unroll
      for (int e = 0; e < 16; ++e) bfr[nt][e] = Bsm[col * BS + kb + e];
    }
#pragma unroll
    for (int mt = 0; mt < 2; ++mt)
#pragma unroll
      for (int nt = 0; nt < 2; ++nt)
        acc[mt][nt] = wmma_bf16(af[mt], bfr[nt], acc[mt][nt]);
  }

#pragma unroll
  for (int mt = 0; mt < 2; ++mt)
#pragma unroll
    for (int nt = 0; nt < 2; ++nt) {
      const int col = n0 + wn * 32 + nt * 16 + l15;
      const float bv = bias[col];
#pragma unroll
      for (int r = 0; r < 8; ++r) {
        const int row = m0 + wm * 32 + mt * 16 + r + halfsel * 8;
        store_out(&Cout[(size_t)row * N + col], acc[mt][nt][r] + bv);
      }
    }
}

// ---------------------------------------------------------------------------
// Flash attention: block = (b, h, 128 query rows); wave = 16-row strip.
// Q/K/V are bf16 [B,T,H*hd]. K tile staged by the Tensor Data Mover with LDS
// row padding; V staged transposed by ALU path. Online softmax, fp32 O acc.
// ---------------------------------------------------------------------------
__global__ __launch_bounds__(256) void flash_attn_wmma(
    const unsigned short* __restrict__ Qg, const unsigned short* __restrict__ Kg,
    const unsigned short* __restrict__ Vg, unsigned short* __restrict__ Yg) {
  constexpr int T = 2048, Cc = 1024, HD = 64, QT = 128, KC = 64, LS = 72;
  __shared__ unsigned short Ks[KC * LS];     // [key][dim]  (TDM-filled, padded)
  __shared__ unsigned short Vt[HD * LS];     // [dim][key]  (transposed)
  __shared__ unsigned short Ps[8 * 16 * LS]; // per-wave P tile [row][key]
  unsigned int* Vt32 = (unsigned int*)Vt;

  const int tid = threadIdx.x;
  const int lane = tid & 31;
  const int w = tid >> 5;
  const int l15 = lane & 15;
  const int halfsel = (lane < 16) ? 0 : 1;
  const int qt = blockIdx.x, h = blockIdx.y, b = blockIdx.z;
  const int q0 = qt * QT + w * 16;  // wave's query strip base
  const size_t base = (size_t)b * T * Cc + (size_t)h * HD;

  // Q fragments (A-layout, K-dim = head dim, 2 frags of 32), pre-scaled
  v16us qf[2];
  {
    const int kb = halfsel * 16;
    const size_t qrow = base + (size_t)(q0 + l15) * Cc;
#pragma unroll
    for (int f = 0; f < 2; ++f)
#pragma unroll
      for (int e = 0; e < 16; ++e) {
        const int d = e + kb + f * 32;
        qf[f][e] = f2bf(bf2f(Qg[qrow + d]) * 0.125f);  // 1/sqrt(64)
      }
  }

  float m_run[8], l_run[8];
#pragma unroll
  for (int r = 0; r < 8; ++r) { m_run[r] = -1e30f; l_run[r] = 0.f; }
  v8f o_acc[4] = {};

  // V transpose staging map: thread -> key pair (2*kp, 2*kp+1) x 8 dims
  const int kp = tid >> 3;           // 0..31
  const int vdoff = (tid & 7) * 8;   // 0..56
  unsigned short* pw = &Ps[w * 16 * LS];
  const unsigned ks_lds = lds_off(&Ks[0]);

  // TDM descriptor group1 (loop-invariant): bf16, 64x64 tile, padded rows
  // row = 64 elem = 32 DW -> pad_interval code 4; pad 4 DW (8 halfwords) -> code 3
  v8i g1;
  g1[0] = (1 << 16) | (1 << 20) | (4 << 22) | (3 << 25);
  g1[1] = 64 << 16;      // tensor_dim0 = 64   (bits 79:48)
  g1[2] = 2048 << 16;    // tensor_dim1 = 2048 (bits 111:80)
  g1[3] = 64 << 16;      // tile_dim0 = 64     (bits 127:112)
  g1[4] = 64;            // tile_dim1 = 64     (bits 143:128)
  g1[5] = 1024;          // tensor_dim0_stride = 1024 (bits 207:160)
  g1[6] = 0;
  g1[7] = 0;

  for (int kc = 0; kc < (qt + 1) * QT; kc += KC) {
    __syncthreads();
    if (w == 0) {  // wave 0: issue TDM load of K tile -> Ks
      const unsigned long long ga =
          (unsigned long long)(const void*)(Kg + base + (size_t)kc * Cc);
      v4u g0;
      g0[0] = 1u;                                   // count=1, user mode
      g0[1] = ks_lds;                               // lds_addr
      g0[2] = (unsigned int)ga;                     // global_addr lo
      g0[3] = (unsigned int)((ga >> 32) & 0x01FFFFFFu) | (2u << 30);  // hi|type=2
      tdm_load_2d(g0, g1);
    }
    {  // all threads: stage V transposed [dim][key], packed key pairs
      const size_t vp = base + (size_t)(kc + 2 * kp) * Cc + vdoff;
#pragma unroll
      for (int j = 0; j < 8; ++j) {
        const unsigned int lo = Vg[vp + j];
        const unsigned int hi = Vg[vp + Cc + j];
        Vt32[(vdoff + j) * (LS / 2) + kp] = lo | (hi << 16);
      }
    }
    if (w == 0) __builtin_amdgcn_s_wait_tensorcnt(0);
    __syncthreads();

    // S = Q * K^T : 4 key-tiles of 16, K-dim = 64 (2 WMMAs each)
    v8f s[4] = {};
#pragma unroll
    for (int nt = 0; nt < 4; ++nt) {
      v16us b0, b1;
      const int key = nt * 16 + l15;
      const int kb = halfsel * 16;
#pragma unroll
      for (int e = 0; e < 16; ++e) {
        b0[e] = Ks[key * LS + kb + e];
        b1[e] = Ks[key * LS + 32 + kb + e];
      }
      s[nt] = wmma_bf16(qf[0], b0, s[nt]);
      s[nt] = wmma_bf16(qf[1], b1, s[nt]);
    }

    // causal mask (only chunks that can cross the diagonal of this strip)
    if (kc + KC > q0) {
#pragma unroll
      for (int nt = 0; nt < 4; ++nt) {
        const int colg = kc + nt * 16 + l15;
#pragma unroll
        for (int r = 0; r < 8; ++r)
          if (colg > q0 + r + halfsel * 8) s[nt][r] = -1e30f;
      }
    }

    // online softmax per row (row r lives in VGPR r across a 16-lane half)
#pragma unroll
    for (int r = 0; r < 8; ++r) {
      float mx = fmaxf(fmaxf(s[0][r], s[1][r]), fmaxf(s[2][r], s[3][r]));
#pragma unroll
      for (int off = 1; off < 16; off <<= 1)
        mx = fmaxf(mx, __shfl_xor(mx, off, 32));
      const float mn = fmaxf(m_run[r], mx);
      const float alpha = __expf(m_run[r] - mn);
      float ls = 0.f;
#pragma unroll
      for (int nt = 0; nt < 4; ++nt) {
        const float p = __expf(s[nt][r] - mn);
        s[nt][r] = p;
        ls += p;
      }
#pragma unroll
      for (int off = 1; off < 16; off <<= 1) ls += __shfl_xor(ls, off, 32);
      l_run[r] = l_run[r] * alpha + ls;
      m_run[r] = mn;
#pragma unroll
      for (int dt = 0; dt < 4; ++dt) o_acc[dt][r] *= alpha;
    }

    // C-layout -> A-layout via per-wave LDS round trip
#pragma unroll
    for (int nt = 0; nt < 4; ++nt) {
      const int col = nt * 16 + l15;
#pragma unroll
      for (int r = 0; r < 8; ++r)
        pw[(r + halfsel * 8) * LS + col] = f2bf(s[nt][r]);
    }

    v16us pf[2];
    {
      const int kb = halfsel * 16;
#pragma unroll
      for (int e = 0; e < 16; ++e) {
        const int kk = e + kb;
        pf[0][e] = pw[l15 * LS + kk];
        pf[1][e] = pw[l15 * LS + kk + 32];
      }
    }

    // O += P * V : 4 dim-tiles of 16, K-dim = 64 keys (2 WMMAs each)
#pragma unroll
    for (int dt = 0; dt < 4; ++dt) {
      v16us b0, b1;
      const int dim = dt * 16 + l15;
      const int kb = halfsel * 16;
#pragma unroll
      for (int e = 0; e < 16; ++e) {
        b0[e] = Vt[dim * LS + kb + e];
        b1[e] = Vt[dim * LS + 32 + kb + e];
      }
      o_acc[dt] = wmma_bf16(pf[0], b0, o_acc[dt]);
      o_acc[dt] = wmma_bf16(pf[1], b1, o_acc[dt]);
    }
  }

  // normalize and write Y (bf16)
#pragma unroll
  for (int dt = 0; dt < 4; ++dt) {
    const int dim = dt * 16 + l15;
#pragma unroll
    for (int r = 0; r < 8; ++r) {
      const int rowg = q0 + r + halfsel * 8;
      Yg[base + (size_t)rowg * Cc + dim] = f2bf(o_acc[dt][r] / l_run[r]);
    }
  }
}

// ---------------------------------------------------------------------------
extern "C" void kernel_launch(void* const* d_in, const int* in_sizes, int n_in,
                              void* d_out, int out_size, void* d_ws, size_t ws_size,
                              hipStream_t stream) {
  (void)in_sizes; (void)n_in; (void)out_size; (void)ws_size;
  const float* x  = (const float*)d_in[0];
  const float* Wq = (const float*)d_in[1];
  const float* bq = (const float*)d_in[2];
  const float* Wk = (const float*)d_in[3];
  const float* bk = (const float*)d_in[4];
  const float* Wv = (const float*)d_in[5];
  const float* bv = (const float*)d_in[6];
  const float* Wp = (const float*)d_in[7];
  const float* bp = (const float*)d_in[8];

  const int Bb = 2, T = 2048, Cc = 1024;
  const int M = Bb * T, N = Cc, K = Cc;

  const size_t elems = (size_t)M * N;
  unsigned short* Q  = (unsigned short*)d_ws;   // bf16 scratch
  unsigned short* Kq = Q + elems;
  unsigned short* V  = Kq + elems;
  unsigned short* Y  = V + elems;               // total 32 MB of ws

  dim3 gg(N / 64, M / 128);
  gemm_bias_wmma<float, unsigned short><<<gg, 256, 0, stream>>>(x, Wq, bq, Q,  M, N, K);
  gemm_bias_wmma<float, unsigned short><<<gg, 256, 0, stream>>>(x, Wk, bk, Kq, M, N, K);
  gemm_bias_wmma<float, unsigned short><<<gg, 256, 0, stream>>>(x, Wv, bv, V,  M, N, K);
  flash_attn_wmma<<<dim3(T / 128, 16, Bb), 256, 0, stream>>>(Q, Kq, V, Y);
  gemm_bias_wmma<unsigned short, float><<<gg, 256, 0, stream>>>(Y, Wp, bp, (float*)d_out, M, N, K);
}
